// Model_68968584839622
// MI455X (gfx1250) — compile-verified, matrix operand order checked by reading, not run on hardware
//
#include <hip/hip_runtime.h>
#include <hip/hip_bf16.h>

// ---------------------------------------------------------------------------
// Soft decision tree (depth 5), fused on gfx1250 (MI455X):
//   K1: gates + path  (B x 32)                        -> d_ws[0, 2MB)
//   K2: W1 f32 [l][d][h] -> bf16 transposed [l][h][d] -> d_ws[2MB, 10.4MB)
//   K3: per (64-row tile, leaf): bf16 WMMA GEMM (512->256) + relu,
//       tiny head (256->3), softmax, path-weighted accumulate -> out (B x 3)
// B fragments stream straight from L2 in WMMA register layout (no LDS, no
// barriers in the K loop); A fragments come from a bf16 x tile in LDS.
// ---------------------------------------------------------------------------

typedef __attribute__((ext_vector_type(16))) __bf16 v16bf;
typedef __attribute__((ext_vector_type(8)))  float  v8f;

#define B_TOT   16384
#define DDIM    512
#define HDIM    256
#define NLEAF   32
#define NNODE   31
#define BM      64            // rows per block

#define XSTRIDE 520           // 512 + 8 pad (ushorts): conflict-free A reads
#define HSTRIDE 264           // 256 + 8 pad (ushorts)

__device__ __forceinline__ unsigned short f2bf(float f) {
  unsigned int u = __float_as_uint(f);
  u += 0x7FFFu + ((u >> 16) & 1u);          // round-to-nearest-even
  return (unsigned short)(u >> 16);
}
__device__ __forceinline__ float bf2f(unsigned short s) {
  return __uint_as_float(((unsigned int)s) << 16);
}

// ------------------------- Kernel 1: gates + path --------------------------
__global__ void __launch_bounds__(256)
path_kernel(const float* __restrict__ x, const float* __restrict__ Wg,
            const float* __restrict__ bg, float* __restrict__ path_out) {
  __shared__ float wg_s[NNODE * DDIM];                 // 62 KB
  const int tid = threadIdx.x;
  for (int e = tid; e < NNODE * DDIM; e += 256) wg_s[e] = Wg[e];
  __syncthreads();

  const int row = blockIdx.x * 256 + tid;
  float acc[NNODE];
#pragma unroll
  for (int n = 0; n < NNODE; ++n) acc[n] = 0.f;

  const float* xr = x + (size_t)row * DDIM;
  for (int d = 0; d < DDIM; d += 4) {
    const float4 xv = *(const float4*)(xr + d);
#pragma unroll
    for (int n = 0; n < NNODE; ++n) {
      const float4 wv = *(const float4*)(wg_s + n * DDIM + d);
      acc[n] += xv.x * wv.x + xv.y * wv.y + xv.z * wv.z + xv.w * wv.w;
    }
  }
  float gate[NNODE];
#pragma unroll
  for (int n = 0; n < NNODE; ++n)
    gate[n] = 1.f / (1.f + __expf(-2.0f * (acc[n] + bg[n])));  // /TEMP == *2

#pragma unroll
  for (int l = 0; l < NLEAF; ++l) {
    float p = 1.f;
#pragma unroll
    for (int d = 0; d < 5; ++d) {
      const int node = (1 << d) - 1 + (l >> (5 - d));
      const float g = gate[node];
      p *= ((l >> (4 - d)) & 1) ? (1.f - g) : g;
    }
    path_out[(size_t)row * NLEAF + l] = p;
  }
}

// -------------- Kernel 2: W1 -> bf16, transpose to [l][h][d] ---------------
// grid = NLEAF * (DDIM/64) * (HDIM/64) blocks of 256 threads; 64x64 tiles.
__global__ void __launch_bounds__(256)
w1t_kernel(const float* __restrict__ W1, unsigned short* __restrict__ W1T) {
  __shared__ unsigned short t_lds[64 * 66];            // pad 2 -> conflict-free
  const int tid  = threadIdx.x;
  const int bid  = blockIdx.x;
  const int l    = bid >> 5;                           // 32 tiles per leaf
  const int tile = bid & 31;
  const int d0   = (tile >> 2) * 64;                   // 8 d-tiles
  const int h0   = (tile & 3) * 64;                    // 4 h-tiles

  const float* src = W1 + ((size_t)l * DDIM) * HDIM;
  for (int e = tid; e < 64 * 64; e += 256) {
    const int r = e >> 6, c = e & 63;                  // r: d-off, c: h-off
    t_lds[r * 66 + c] = f2bf(src[(size_t)(d0 + r) * HDIM + (h0 + c)]);
  }
  __syncthreads();
  unsigned short* dst = W1T + ((size_t)l * HDIM) * DDIM;
  for (int e = tid; e < 64 * 64; e += 256) {
    const int r = e >> 6, c = e & 63;                  // r: h-off, c: d-off
    dst[(size_t)(h0 + r) * DDIM + (d0 + c)] = t_lds[c * 66 + r];
  }
}

// --------------------- Kernel 3: fused expert evaluation -------------------
__global__ void __launch_bounds__(256)
moe_kernel(const float* __restrict__ x,  const unsigned short* __restrict__ W1T,
           const float* __restrict__ b1, const float* __restrict__ W2,
           const float* __restrict__ b2, const float* __restrict__ path,
           float* __restrict__ out) {
  __shared__ unsigned short x_lds[BM * XSTRIDE];       // 66560 B, bf16 x tile
  __shared__ unsigned short h_lds[BM * HSTRIDE];       // 33792 B, bf16 h tile

  const int tid  = threadIdx.x;
  const int lane = tid & 31;
  const int wave = tid >> 5;
  const int rowGroup = wave >> 1;       // 4 x 16-row slabs
  const int nGroup   = wave & 1;        // 2 x 128-col slabs
  const int rowBase  = blockIdx.x * BM;
  const int half = lane >> 4;           // ISA 16-bit frag layout: K-half select
  const int l16  = lane & 15;

  // stage x tile (64 x 512) as bf16, coalesced
  for (int e = tid; e < BM * DDIM; e += 256) {
    const int r = e >> 9, c = e & 511;
    x_lds[r * XSTRIDE + c] = f2bf(x[(size_t)(rowBase + r) * DDIM + c]);
  }
  __syncthreads();

  float o0 = 0.f, o1 = 0.f, o2 = 0.f;   // per-row output accum (threads 0..63)

#pragma unroll 1
  for (int l = 0; l < NLEAF; ++l) {
    v8f acc[8] = {};
    const unsigned short* w1tl = W1T + (size_t)l * HDIM * DDIM;

#pragma unroll 1
    for (int kc = 0; kc < DDIM; kc += 32) {
      // A fragment (16x32): lanes 0-15 K={0..7,16..23}, lanes16-31 K={8..15,24..31}
      union { v16bf v; uint4 q[2]; } af;
      const unsigned short* xr = &x_lds[(rowGroup * 16 + l16) * XSTRIDE + kc];
      af.q[0] = *(const uint4*)(xr + 8 * half);
      af.q[1] = *(const uint4*)(xr + 16 + 8 * half);

      // B fragments straight from L2: lane -> column h, K contiguous in [l][h][d]
      uint4 bq[16];
#pragma unroll
      for (int t = 0; t < 8; ++t) {
        const unsigned short* wr =
            w1tl + (size_t)(nGroup * 128 + t * 16 + l16) * DDIM + kc;
        bq[2 * t]     = *(const uint4*)(wr + 8 * half);
        bq[2 * t + 1] = *(const uint4*)(wr + 16 + 8 * half);
      }
#pragma unroll
      for (int t = 0; t < 8; ++t) {
        union { v16bf v; uint4 q[2]; } bfr;
        bfr.q[0] = bq[2 * t];
        bfr.q[1] = bq[2 * t + 1];
        acc[t] = __builtin_amdgcn_wmma_f32_16x16x32_bf16(
            false, af.v, false, bfr.v, (short)0, acc[t], false, false);
      }
    }

    __syncthreads();   // previous leaf's head phase has finished reading h_lds

    // h = relu(acc + b1) -> h_lds[row*HSTRIDE + col] (bf16)
    // C layout: VGPR i -> M = (lane>=16 ? 8 : 0) + i, N = lane&15
#pragma unroll
    for (int t = 0; t < 8; ++t) {
      const int col   = nGroup * 128 + t * 16 + l16;
      const float bias = b1[l * HDIM + col];
#pragma unroll
      for (int i = 0; i < 8; ++i) {
        const int r = rowGroup * 16 + half * 8 + i;
        const float v = acc[t][i] + bias;
        h_lds[r * HSTRIDE + col] = f2bf(v > 0.f ? v : 0.f);
      }
    }
    __syncthreads();

    // head: logits (row x 3), softmax, path-weighted accumulate
    if (tid < BM) {
      const float* w2l = W2 + l * HDIM * 3;
      float l0 = b2[l * 3 + 0], l1 = b2[l * 3 + 1], l2 = b2[l * 3 + 2];
      const unsigned short* hr = &h_lds[tid * HSTRIDE];
      for (int h = 0; h < HDIM; ++h) {
        const float hv = bf2f(hr[h]);
        l0 += hv * w2l[h * 3 + 0];
        l1 += hv * w2l[h * 3 + 1];
        l2 += hv * w2l[h * 3 + 2];
      }
      const float m  = fmaxf(l0, fmaxf(l1, l2));
      const float e0 = __expf(l0 - m), e1 = __expf(l1 - m), e2 = __expf(l2 - m);
      const float inv = 1.f / (e0 + e1 + e2);
      const float pv  = path[(size_t)(rowBase + tid) * NLEAF + l];
      o0 += pv * e0 * inv;
      o1 += pv * e1 * inv;
      o2 += pv * e2 * inv;
    }
  }

  if (tid < BM) {
    const size_t r = (size_t)(rowBase + tid);
    out[r * 3 + 0] = o0;
    out[r * 3 + 1] = o1;
    out[r * 3 + 2] = o2;
  }
}

// --------------------------------- launch ----------------------------------
extern "C" void kernel_launch(void* const* d_in, const int* in_sizes, int n_in,
                              void* d_out, int out_size, void* d_ws, size_t ws_size,
                              hipStream_t stream) {
  (void)in_sizes; (void)n_in; (void)out_size; (void)ws_size;
  const float* x  = (const float*)d_in[0];
  const float* Wg = (const float*)d_in[1];
  const float* bg = (const float*)d_in[2];
  const float* W1 = (const float*)d_in[3];
  const float* b1 = (const float*)d_in[4];
  const float* W2 = (const float*)d_in[5];
  const float* b2 = (const float*)d_in[6];
  float* out = (float*)d_out;

  float*          path = (float*)d_ws;                        // 2 MB
  unsigned short* w1t  = (unsigned short*)((char*)d_ws + (size_t)B_TOT * NLEAF * 4);
                                                              // 8.4 MB bf16

  path_kernel<<<B_TOT / 256, 256, 0, stream>>>(x, Wg, bg, path);
  w1t_kernel<<<NLEAF * (DDIM / 64) * (HDIM / 64), 256, 0, stream>>>(W1, w1t);
  moe_kernel<<<B_TOT / BM, 256, 0, stream>>>(x, w1t, b1, W2, b2, path, out);
}
